// MoE_16655883174694
// MI455X (gfx1250) — compile-verified, matrix operand order checked by reading, not run on hardware
//
#include <hip/hip_runtime.h>
#include <hip/hip_bf16.h>

// ---------------------------------------------------------------------------
// MoE forward for MI455X (gfx1250, wave32, WMMA).
// Gather-based: only routed (token, expert) pairs are computed (1/4 of dense).
// bf16 WMMA (f32 accumulate); f32 weights converted to bf16 via v_perm packs
// while staging into double-buffered LDS. Per-expert weight slices (64 MB)
// are L2-resident (192 MB L2), so tile re-reads are cheap after first touch.
// ---------------------------------------------------------------------------

typedef __attribute__((ext_vector_type(16))) __bf16 v16bf;
typedef __attribute__((ext_vector_type(8)))  float  v8f;

#define BTOK 8192   // B*T tokens
#define DDIM 2048
#define EEXP 8
#define HDIM 8192
#define TOTROWS (BTOK * 2)   // each token goes to exactly K=2 experts

// workspace layout (bytes):
//   [0)      counts   : int[8]
//   [64)     offsets  : int[9]
//   [1024)   row_ids  : int[E*BT]      (256 KB)
//   [+256KB) row_gates: float[E*BT]    (256 KB)
//   [1MB)    Hbuf     : bf16[TOTROWS * HDIM]  (256 MB)

__device__ __forceinline__ unsigned int bfround(unsigned int u) {
  return u + 0x7fffu + ((u >> 16) & 1u);   // round-to-nearest-even for bf16
}
// pack two f32 -> packed bf16x2 with one v_perm_b32 (takes bytes [3:2] of each)
__device__ __forceinline__ unsigned int pack2(float a, float b) {
  union { float f; unsigned int u; } ua, ub; ua.f = a; ub.f = b;
  return __builtin_amdgcn_perm(bfround(ub.u), bfround(ua.u), 0x07060302u);
}
__device__ __forceinline__ unsigned short f2bf(float f) {
  union { float f; unsigned int u; } v; v.f = f;
  return (unsigned short)(bfround(v.u) >> 16);
}

union Frag { unsigned int u[8]; v16bf v; };

// ---------------------------------------------------------------------------
// init: zero output accumulator and per-expert counters
// ---------------------------------------------------------------------------
__global__ void moe_init_kernel(float* __restrict__ out, int n, int* __restrict__ counts) {
  int i = blockIdx.x * blockDim.x + threadIdx.x;
  if (i < n) out[i] = 0.0f;
  if (i < EEXP) counts[i] = 0;
}

// ---------------------------------------------------------------------------
// router: noisy top-2 gating + append (token, gate) to per-expert lists.
// ---------------------------------------------------------------------------
__global__ __launch_bounds__(64) void moe_router_kernel(
    const float* __restrict__ x, const float* __restrict__ eps,
    const float* __restrict__ Wr, const float* __restrict__ br,
    const float* __restrict__ Wn, const float* __restrict__ bn,
    int* __restrict__ counts, int* __restrict__ row_ids, float* __restrict__ row_gates) {
  __shared__ float red[64 * 16];
  __shared__ float sums[16];
  const int t = blockIdx.x;
  const int tid = threadIdx.x;
  float pr[EEXP] = {}, pn[EEXP] = {};
  const float* xr = x + (size_t)t * DDIM;
  for (int d = tid; d < DDIM; d += 64) {
    float xv = xr[d];
#pragma unroll
    for (int e = 0; e < EEXP; ++e) {
      pr[e] += xv * Wr[d * EEXP + e];
      pn[e] += xv * Wn[d * EEXP + e];
    }
  }
#pragma unroll
  for (int e = 0; e < EEXP; ++e) { red[tid * 16 + e] = pr[e]; red[tid * 16 + 8 + e] = pn[e]; }
  __syncthreads();
  if (tid < 16) {
    float s = 0.f;
    for (int i = 0; i < 64; ++i) s += red[i * 16 + tid];
    sums[tid] = s;
  }
  __syncthreads();
  if (tid == 0) {
    float noisy[EEXP];
#pragma unroll
    for (int e = 0; e < EEXP; ++e) {
      float lg = sums[e] + br[e];
      float nl = sums[8 + e] + bn[e];
      float sp = fmaxf(nl, 0.f) + log1pf(expf(-fabsf(nl)));  // stable softplus
      noisy[e] = lg + eps[(size_t)t * EEXP + e] * sp;
    }
    int i1 = 0;
    for (int e = 1; e < EEXP; ++e) if (noisy[e] > noisy[i1]) i1 = e;
    int i2 = (i1 == 0) ? 1 : 0;
    for (int e = 0; e < EEXP; ++e) if (e != i1 && noisy[e] > noisy[i2]) i2 = e;
    float e2 = expf(noisy[i2] - noisy[i1]);
    float inv = 1.0f / (1.0f + e2);
    int s1 = atomicAdd(&counts[i1], 1);
    row_ids[i1 * BTOK + s1] = t; row_gates[i1 * BTOK + s1] = inv;
    int s2 = atomicAdd(&counts[i2], 1);
    row_ids[i2 * BTOK + s2] = t; row_gates[i2 * BTOK + s2] = e2 * inv;
  }
}

__global__ void moe_prefix_kernel(const int* __restrict__ counts, int* __restrict__ offsets) {
  if (blockIdx.x == 0 && threadIdx.x == 0) {
    int s = 0;
    for (int e = 0; e < EEXP; ++e) { offsets[e] = s; s += counts[e]; }
    offsets[EEXP] = s;
  }
}

// ---------------------------------------------------------------------------
// GEMM geometry: 128x128 C tile, 256 threads = 8 waves; wave w covers
// M [ (w&3)*32, +32 ) x N [ (w>>2)*64, +64 ) => 2x4 WMMA accumulators.
// LDS tiles: row stride 17 dwords (34 bf16); gcd(17,64)=1 -> conflict-free
// fragment reads. Double-buffered: global loads for tile k+1 issue before
// the WMMAs of tile k; packed stores land in the alternate buffer.
//
// bf16 A fragment (16x32, ISA 7.12.2): lanes 0-15 row=lane, K 0..7 / 16..23;
// lanes 16-31 K 8..15 / 24..31. B staged transposed [n][k] so B fragments
// use identical addressing.
// ---------------------------------------------------------------------------

// GEMM1: Hbuf[ofs+slot][h] = relu( x[row_ids[slot]] @ W1[e] + b1[e] )  (bf16)
__global__ __launch_bounds__(256) void moe_gemm1_kernel(
    const float* __restrict__ x, const float* __restrict__ W1, const float* __restrict__ b1,
    const int* __restrict__ counts, const int* __restrict__ offsets,
    const int* __restrict__ row_ids, unsigned short* __restrict__ Hbuf) {
  __shared__ unsigned int lsA[2][128 * 17];
  __shared__ unsigned int lsB[2][128 * 17];

  const int e = blockIdx.z;
  const int cnt = counts[e];
  const int rowTile = blockIdx.y;
  if (rowTile * 128 >= cnt) return;  // uniform early exit (empty tile)
  const int n0 = blockIdx.x * 128;

  const int tid = threadIdx.x;
  const int lane = tid & 31, wave = tid >> 5;
  const int half = lane >> 4, r16 = lane & 15;
  const int mb = (wave & 3) * 32;
  const int nb = (wave >> 2) * 64;
  const int kb2 = half * 4;  // dword offset of per-lane K slice

  // ---- K-invariant staging addresses (hoisted out of the K loop) ----
  const float* aptr[4]; int adst[4];
#pragma unroll
  for (int it = 0; it < 4; ++it) {
    int idx = tid + it * 256;
    int row = idx >> 3;
    int kc = (idx & 7) * 4;
    int slot = rowTile * 128 + row;
    int trow = (slot < cnt) ? row_ids[e * BTOK + slot] : 0;
    aptr[it] = x + (size_t)trow * DDIM + kc;
    adst[it] = row * 17 + (kc >> 1);
  }
  const float* bptr0[4]; const float* bptr1[4]; int bdst[4];
#pragma unroll
  for (int it = 0; it < 4; ++it) {
    int idx = tid + it * 256;
    int kp = idx >> 6;        // k-pair 0..15
    int np = idx & 63;        // n-pair 0..63
    const float* base = W1 + ((size_t)e * DDIM + 2 * kp) * HDIM + n0 + 2 * np;
    bptr0[it] = base;
    bptr1[it] = base + HDIM;
    bdst[it] = (2 * np) * 17 + kp;
  }

  v8f acc[2][4] = {};
  float4 la[4]; float2 lb0[4], lb1[4];

  // prologue: stage tile 0 into buffer 0
#pragma unroll
  for (int it = 0; it < 4; ++it) {
    la[it] = *(const float4*)(aptr[it]);
    lb0[it] = *(const float2*)(bptr0[it]);
    lb1[it] = *(const float2*)(bptr1[it]);
  }
#pragma unroll
  for (int it = 0; it < 4; ++it) {
    lsA[0][adst[it]]     = pack2(la[it].x, la[it].y);
    lsA[0][adst[it] + 1] = pack2(la[it].z, la[it].w);
    lsB[0][bdst[it]]      = pack2(lb0[it].x, lb1[it].x);
    lsB[0][bdst[it] + 17] = pack2(lb0[it].y, lb1[it].y);
  }

  const int nIter = DDIM / 32;
  for (int ki = 0; ki < nIter; ++ki) {
    const int buf = ki & 1;
    __syncthreads();
    // issue global loads for tile ki+1 (overlap with WMMAs below)
    if (ki + 1 < nIter) {
      int k0 = (ki + 1) * 32;
#pragma unroll
      for (int it = 0; it < 4; ++it) {
        la[it] = *(const float4*)(aptr[it] + k0);
        lb0[it] = *(const float2*)(bptr0[it] + (size_t)k0 * HDIM);
        lb1[it] = *(const float2*)(bptr1[it] + (size_t)k0 * HDIM);
      }
      if (ki + 2 < nIter)  // global_prefetch of weight tile two steps ahead
        __builtin_prefetch(bptr0[0] + (size_t)(ki + 2) * 32 * HDIM, 0, 0);
    }
    // compute tile ki from LDS buffer `buf`
    Frag aF[2], bF[4];
#pragma unroll
    for (int mt = 0; mt < 2; ++mt) {
      const unsigned int* p = &lsA[buf][(mb + mt * 16 + r16) * 17 + kb2];
#pragma unroll
      for (int i = 0; i < 4; ++i) { aF[mt].u[i] = p[i]; aF[mt].u[4 + i] = p[8 + i]; }
    }
#pragma unroll
    for (int nt = 0; nt < 4; ++nt) {
      const unsigned int* p = &lsB[buf][(nb + nt * 16 + r16) * 17 + kb2];
#pragma unroll
      for (int i = 0; i < 4; ++i) { bF[nt].u[i] = p[i]; bF[nt].u[4 + i] = p[8 + i]; }
    }
#pragma unroll
    for (int mt = 0; mt < 2; ++mt)
#pragma unroll
      for (int nt = 0; nt < 4; ++nt)
        acc[mt][nt] = __builtin_amdgcn_wmma_f32_16x16x32_bf16(
            false, aF[mt].v, false, bF[nt].v, (short)0, acc[mt][nt], false, false);
    // stage tile ki+1 into the alternate buffer
    if (ki + 1 < nIter) {
      unsigned int* A = lsA[buf ^ 1];
      unsigned int* B = lsB[buf ^ 1];
#pragma unroll
      for (int it = 0; it < 4; ++it) {
        A[adst[it]]     = pack2(la[it].x, la[it].y);
        A[adst[it] + 1] = pack2(la[it].z, la[it].w);
        B[bdst[it]]      = pack2(lb0[it].x, lb1[it].x);
        B[bdst[it] + 17] = pack2(lb0[it].y, lb1[it].y);
      }
    }
  }

  // epilogue: bias + ReLU -> bf16 Hbuf rows
  const int ofs = offsets[e];
  float bias[4];
#pragma unroll
  for (int nt = 0; nt < 4; ++nt) bias[nt] = b1[e * HDIM + n0 + nb + nt * 16 + r16];
#pragma unroll
  for (int mt = 0; mt < 2; ++mt) {
#pragma unroll
    for (int v = 0; v < 8; ++v) {
      int slot = rowTile * 128 + mb + mt * 16 + half * 8 + v;
      if (slot < cnt) {
        size_t rb = (size_t)(ofs + slot) * HDIM + n0 + nb + r16;
#pragma unroll
        for (int nt = 0; nt < 4; ++nt)
          Hbuf[rb + nt * 16] = f2bf(fmaxf(acc[mt][nt][v] + bias[nt], 0.f));
      }
    }
  }
}

// GEMM2: out[t] += gate * (Hbuf_row @ W2[e] + b2[e])   (atomic f32 scatter)
__global__ __launch_bounds__(256) void moe_gemm2_kernel(
    const float* __restrict__ W2, const float* __restrict__ b2,
    const int* __restrict__ counts, const int* __restrict__ offsets,
    const int* __restrict__ row_ids, const float* __restrict__ row_gates,
    const unsigned short* __restrict__ Hbuf, float* __restrict__ out) {
  __shared__ unsigned int lsA[2][128 * 17];
  __shared__ unsigned int lsB[2][128 * 17];

  const int e = blockIdx.z;
  const int cnt = counts[e];
  const int rowTile = blockIdx.y;
  if (rowTile * 128 >= cnt) return;
  const int d0 = blockIdx.x * 128;
  const int ofs = offsets[e];

  const int tid = threadIdx.x;
  const int lane = tid & 31, wave = tid >> 5;
  const int half = lane >> 4, r16 = lane & 15;
  const int mb = (wave & 3) * 32;
  const int nb = (wave >> 2) * 64;
  const int kb2 = half * 4;

  // K-invariant staging addresses
  const unsigned int* HbufU = (const unsigned int*)Hbuf;
  const unsigned int* aptr[2]; int adst[2];
#pragma unroll
  for (int it = 0; it < 2; ++it) {
    int idx = tid + it * 256;
    int row = idx >> 2;
    int uc = (idx & 3) * 4;
    int slot = rowTile * 128 + row;
    int gr = ofs + ((slot < cnt) ? slot : 0);
    aptr[it] = HbufU + (size_t)gr * (HDIM / 2) + uc;
    adst[it] = row * 17 + uc;
  }
  const float* bptr0[4]; const float* bptr1[4]; int bdst[4];
#pragma unroll
  for (int it = 0; it < 4; ++it) {
    int idx = tid + it * 256;
    int kp = idx >> 6;
    int np = idx & 63;
    const float* base = W2 + ((size_t)e * HDIM + 2 * kp) * DDIM + d0 + 2 * np;
    bptr0[it] = base;
    bptr1[it] = base + DDIM;
    bdst[it] = (2 * np) * 17 + kp;
  }

  v8f acc[2][4] = {};
  uint4 ua[2]; float2 lb0[4], lb1[4];

  // prologue: stage tile 0 into buffer 0
#pragma unroll
  for (int it = 0; it < 2; ++it) ua[it] = *(const uint4*)(aptr[it]);
#pragma unroll
  for (int it = 0; it < 4; ++it) {
    lb0[it] = *(const float2*)(bptr0[it]);
    lb1[it] = *(const float2*)(bptr1[it]);
  }
#pragma unroll
  for (int it = 0; it < 2; ++it) {
    lsA[0][adst[it] + 0] = ua[it].x; lsA[0][adst[it] + 1] = ua[it].y;
    lsA[0][adst[it] + 2] = ua[it].z; lsA[0][adst[it] + 3] = ua[it].w;
  }
#pragma unroll
  for (int it = 0; it < 4; ++it) {
    lsB[0][bdst[it]]      = pack2(lb0[it].x, lb1[it].x);
    lsB[0][bdst[it] + 17] = pack2(lb0[it].y, lb1[it].y);
  }

  const int nIter = HDIM / 32;
  for (int ki = 0; ki < nIter; ++ki) {
    const int buf = ki & 1;
    __syncthreads();
    if (ki + 1 < nIter) {
      int k0 = (ki + 1) * 32;
#pragma unroll
      for (int it = 0; it < 2; ++it) ua[it] = *(const uint4*)(aptr[it] + (k0 >> 1));
#pragma unroll
      for (int it = 0; it < 4; ++it) {
        lb0[it] = *(const float2*)(bptr0[it] + (size_t)k0 * DDIM);
        lb1[it] = *(const float2*)(bptr1[it] + (size_t)k0 * DDIM);
      }
      if (ki + 2 < nIter)
        __builtin_prefetch(bptr0[0] + (size_t)(ki + 2) * 32 * DDIM, 0, 0);
    }
    Frag aF[2], bF[4];
#pragma unroll
    for (int mt = 0; mt < 2; ++mt) {
      const unsigned int* p = &lsA[buf][(mb + mt * 16 + r16) * 17 + kb2];
#pragma unroll
      for (int i = 0; i < 4; ++i) { aF[mt].u[i] = p[i]; aF[mt].u[4 + i] = p[8 + i]; }
    }
#pragma unroll
    for (int nt = 0; nt < 4; ++nt) {
      const unsigned int* p = &lsB[buf][(nb + nt * 16 + r16) * 17 + kb2];
#pragma unroll
      for (int i = 0; i < 4; ++i) { bF[nt].u[i] = p[i]; bF[nt].u[4 + i] = p[8 + i]; }
    }
#pragma unroll
    for (int mt = 0; mt < 2; ++mt)
#pragma unroll
      for (int nt = 0; nt < 4; ++nt)
        acc[mt][nt] = __builtin_amdgcn_wmma_f32_16x16x32_bf16(
            false, aF[mt].v, false, bF[nt].v, (short)0, acc[mt][nt], false, false);
    if (ki + 1 < nIter) {
      unsigned int* A = lsA[buf ^ 1];
      unsigned int* B = lsB[buf ^ 1];
#pragma unroll
      for (int it = 0; it < 2; ++it) {
        A[adst[it] + 0] = ua[it].x; A[adst[it] + 1] = ua[it].y;
        A[adst[it] + 2] = ua[it].z; A[adst[it] + 3] = ua[it].w;
      }
#pragma unroll
      for (int it = 0; it < 4; ++it) {
        B[bdst[it]]      = pack2(lb0[it].x, lb1[it].x);
        B[bdst[it] + 17] = pack2(lb0[it].y, lb1[it].y);
      }
    }
  }

  // epilogue: bias, gate, commutative atomic scatter (2 adds/element total)
  float bias[4];
#pragma unroll
  for (int nt = 0; nt < 4; ++nt) bias[nt] = b2[e * DDIM + d0 + nb + nt * 16 + r16];
#pragma unroll
  for (int mt = 0; mt < 2; ++mt) {
#pragma unroll
    for (int v = 0; v < 8; ++v) {
      int slot = rowTile * 128 + mb + mt * 16 + half * 8 + v;
      if (slot < cnt) {
        int t = row_ids[e * BTOK + slot];
        float g = row_gates[e * BTOK + slot];
        size_t ob = (size_t)t * DDIM + d0 + nb + r16;
#pragma unroll
        for (int nt = 0; nt < 4; ++nt)
          atomicAdd(out + ob + nt * 16, (acc[mt][nt][v] + bias[nt]) * g);
      }
    }
  }
}

// ---------------------------------------------------------------------------
extern "C" void kernel_launch(void* const* d_in, const int* in_sizes, int n_in,
                              void* d_out, int out_size, void* d_ws, size_t ws_size,
                              hipStream_t stream) {
  const float* x   = (const float*)d_in[0];
  const float* eps = (const float*)d_in[1];
  const float* Wr  = (const float*)d_in[2];
  const float* br  = (const float*)d_in[3];
  const float* Wn  = (const float*)d_in[4];
  const float* bn  = (const float*)d_in[5];
  const float* W1  = (const float*)d_in[6];
  const float* b1  = (const float*)d_in[7];
  const float* W2  = (const float*)d_in[8];
  const float* b2  = (const float*)d_in[9];
  float* out = (float*)d_out;

  char* ws = (char*)d_ws;
  int*   counts    = (int*)ws;
  int*   offsets   = (int*)(ws + 64);
  int*   row_ids   = (int*)(ws + 1024);
  float* row_gates = (float*)(ws + 1024 + sizeof(int) * EEXP * BTOK);
  unsigned short* Hbuf = (unsigned short*)(ws + (1u << 20));  // 256 MB bf16 activations

  moe_init_kernel<<<(out_size + 255) / 256, 256, 0, stream>>>(out, out_size, counts);
  moe_router_kernel<<<BTOK, 64, 0, stream>>>(x, eps, Wr, br, Wn, bn,
                                             counts, row_ids, row_gates);
  moe_prefix_kernel<<<1, 32, 0, stream>>>(counts, offsets);
  moe_gemm1_kernel<<<dim3(HDIM / 128, BTOK / 128, EEXP), 256, 0, stream>>>(
      x, W1, b1, counts, offsets, row_ids, Hbuf);
  moe_gemm2_kernel<<<dim3(DDIM / 128, BTOK / 128, EEXP), 256, 0, stream>>>(
      W2, b2, counts, offsets, row_ids, row_gates, Hbuf, out);
}
